// TPS_SelfAttention_15272903705329
// MI455X (gfx1250) — matrix operand order repaired and checked by reading, not compiled, not run on hardware
//
#include <hip/hip_runtime.h>
#include <hip/hip_bf16.h>

// Sizes from the reference
#define L_  2048
#define B_  4
#define D_  512
#define H_  8
#define DK_ 64
#define LB_ (L_ * B_)

typedef __bf16 bf16_t;
typedef float  v8f   __attribute__((ext_vector_type(8)));
typedef bf16_t v16bf __attribute__((ext_vector_type(16)));
typedef bf16_t v8bf  __attribute__((ext_vector_type(8)));

static __device__ __forceinline__ v8f vzero8() {
    v8f z = {0.f, 0.f, 0.f, 0.f, 0.f, 0.f, 0.f, 0.f};
    return z;
}

// A-fragment (16x32 bf16, MxK): lane owns row m = lane&15.
// element e -> K = kc*32 + (lane<16 ? 0 : 16) + e
// => two contiguous 16B chunks at k0 and k0+8.  Works for global or LDS ptrs.
static __device__ __forceinline__ v16bf load_a_frag(const bf16_t* rowp, int kc, int lane) {
    const int k0 = kc * 32 + ((lane & 16) ? 16 : 0);
    union U { v16bf v; v8bf h[2]; } u;
    u.h[0] = *(const v8bf*)(rowp + k0);
    u.h[1] = *(const v8bf*)(rowp + k0 + 8);
    return u.v;
}

// B-fragment (32x16 bf16, KxN): lane owns column n; element e -> K = kc*32 + (lane>=16?16:0) + e.
// rowp points at the K-major row for this lane's column => one contiguous 32B load.
static __device__ __forceinline__ v16bf load_b_frag(const bf16_t* rowp, int kc, int lane) {
    return *(const v16bf*)(rowp + kc * 32 + ((lane & 16) ? 16 : 0));
}

// ---------------------------------------------------------------------------
// Kernel 0: bulk fp32 -> bf16 downcast (q, k, Wq, Wk).  Pure bandwidth;
// 8 contiguous elements per thread (32B read / 16B write, fully coalesced).
// ---------------------------------------------------------------------------
__global__ __launch_bounds__(256) void cvt_kernel(const float* __restrict__ src,
                                                  bf16_t* __restrict__ dst) {
    const size_t i = ((size_t)blockIdx.x * blockDim.x + threadIdx.x) * 8;
    union { v8bf v; bf16_t e[8]; } u;
#pragma unroll
    for (int j = 0; j < 8; ++j) u.e[j] = (bf16_t)src[i + j];
    *(v8bf*)(dst + i) = u.v;
}

// ---------------------------------------------------------------------------
// Kernel 1: in-projection  Out[i,n] = sum_k X[i,k]*W[n,k] + bias[n] on
// pre-converted bf16 operands, scattered into head-major outh[b][h][l][dk].
// Grid: (LB_/32) blocks of 256 threads (8 waves); each wave: 2 M-tiles x
// 4 N-tiles (32x64 output), B-fragments shared across the two M-tiles.
// ---------------------------------------------------------------------------
__global__ __launch_bounds__(256) void proj_kernel(const bf16_t* __restrict__ xb,
                                                   const bf16_t* __restrict__ Wb,
                                                   const float* __restrict__ bias,
                                                   bf16_t* __restrict__ outh) {
    const int m0   = blockIdx.x * 32;
    const int wave = threadIdx.x >> 5;
    const int lane = threadIdx.x & 31;
    const int col  = lane & 15;
    const int half = (lane & 16) ? 8 : 0;
    const int nbase = wave * 64;

    const bf16_t* arow0 = xb + (size_t)(m0 + col) * D_;
    const bf16_t* arow1 = arow0 + (size_t)16 * D_;

    v8f c[2][4];
#pragma unroll
    for (int u = 0; u < 2; ++u)
#pragma unroll
        for (int t = 0; t < 4; ++t) c[u][t] = vzero8();

    for (int kc = 0; kc < D_ / 32; ++kc) {
        v16bf a0 = load_a_frag(arow0, kc, lane);
        v16bf a1 = load_a_frag(arow1, kc, lane);
#pragma unroll
        for (int t = 0; t < 4; ++t) {
            const int n = nbase + t * 16 + col;
            v16bf bf = load_b_frag(Wb + (size_t)n * D_, kc, lane);
            c[0][t] = __builtin_amdgcn_wmma_f32_16x16x32_bf16(false, a0, false, bf,
                                                              (short)0, c[0][t], false, false);
            c[1][t] = __builtin_amdgcn_wmma_f32_16x16x32_bf16(false, a1, false, bf,
                                                              (short)0, c[1][t], false, false);
        }
    }

#pragma unroll
    for (int t = 0; t < 4; ++t) {
        const int   n      = nbase + t * 16 + col;
        const int   h      = n >> 6;
        const int   dk     = n & 63;
        const float bias_n = bias[n];
#pragma unroll
        for (int u = 0; u < 2; ++u)
#pragma unroll
            for (int r = 0; r < 8; ++r) {
                const int M  = u * 16 + r + half;
                const int i  = m0 + M;           // flat (l*B + b) row index
                const int l  = i >> 2;           // B_ == 4
                const int bb = i & 3;
                outh[((size_t)(bb * H_ + h) * L_ + l) * DK_ + dk] =
                    (bf16_t)(c[u][t][r] + bias_n);
            }
    }
}

// ---------------------------------------------------------------------------
// Kernel 2: v[L,B,D] fp32 -> vt[B,D,L] bf16 so final-GEMM B-fragments are
// contiguous.  Grid (L_/8, B_), block D_ threads.
// ---------------------------------------------------------------------------
__global__ __launch_bounds__(512) void vt_kernel(const float* __restrict__ v,
                                                 bf16_t* __restrict__ vt) {
    const int l0 = blockIdx.x * 8;
    const int b  = blockIdx.y;
    const int n  = threadIdx.x;

    union { v8bf v; bf16_t e[8]; } u;
#pragma unroll
    for (int j = 0; j < 8; ++j)
        u.e[j] = (bf16_t)v[((size_t)(l0 + j) * B_ + b) * D_ + n];
    *(v8bf*)(vt + ((size_t)(b * D_ + n)) * L_ + l0) = u.v;
}

// ---------------------------------------------------------------------------
// Kernel 3: fused attention.  One workgroup = (batch b, 16-row q-tile).
// 8 waves; wave w owns score columns [w*256, w*256+256) (16 WMMA tiles).
// LDS (dynamic, ~195 KB of the WGP's 320 KB), strides padded so both the
// accumulate stage and the A-fragment ds_load_b128s are bank-conflict-free.
// ---------------------------------------------------------------------------
#define ACC_STRIDE 2052
#define AWB_STRIDE 2056
#define SMEM_BYTES (16 * ACC_STRIDE * 4 + 16 * AWB_STRIDE * 2 + 2048)

__global__ __launch_bounds__(256) void attn_kernel(const bf16_t* __restrict__ Qh,
                                                   const bf16_t* __restrict__ Kh,
                                                   const bf16_t* __restrict__ vt,
                                                   float* __restrict__ outp,
                                                   float* __restrict__ weightedp) {
    extern __shared__ char smem[];
    float*  acc    = (float*)smem;
    bf16_t* awb    = (bf16_t*)(smem + 16 * ACC_STRIDE * 4);
    float*  red    = (float*)(smem + 16 * ACC_STRIDE * 4 + 16 * AWB_STRIDE * 2);
    float*  redmax = red;          // [8][16] per-wave row maxima (reused [16][16] later)
    float*  redsum = red + 128;    // [8][16] per-wave row sums
    float*  rowmax = red + 256;    // [16]
    float*  rowsum = red + 272;    // [16]

    const int b    = blockIdx.y;
    const int q0   = blockIdx.x * 16;
    const int tid  = threadIdx.x;
    const int wave = tid >> 5;
    const int lane = tid & 31;
    const int col  = lane & 15;
    const int half = (lane & 16) ? 8 : 0;
    const float scale = 0.125f;    // 1/sqrt(DK)

    for (int i = tid; i < 16 * ACC_STRIDE; i += 256) acc[i] = 0.f;
    __syncthreads();

    // ---- per-head scores + softmax, accumulated into acc ----
    for (int h = 0; h < H_; ++h) {
        const bf16_t* Qbase = Qh + (size_t)(b * H_ + h) * L_ * DK_;
        const bf16_t* Kbase = Kh + (size_t)(b * H_ + h) * L_ * DK_;
        if (h + 1 < H_) __builtin_prefetch(Kbase + (size_t)L_ * DK_, 0, 1);

        const bf16_t* arow = Qbase + (size_t)(q0 + col) * DK_;
        v16bf a0 = load_a_frag(arow, 0, lane);
        v16bf a1 = load_a_frag(arow, 1, lane);

        v8f c[16];
#pragma unroll
        for (int t = 0; t < 16; ++t) {
            const int     n    = wave * 256 + t * 16 + col;
            const bf16_t* brow = Kbase + (size_t)n * DK_;
            v16bf b0 = load_b_frag(brow, 0, lane);
            v16bf b1 = load_b_frag(brow, 1, lane);
            v8f z = vzero8();
            z    = __builtin_amdgcn_wmma_f32_16x16x32_bf16(false, a0, false, b0,
                                                           (short)0, z, false, false);
            c[t] = __builtin_amdgcn_wmma_f32_16x16x32_bf16(false, a1, false, b1,
                                                           (short)0, z, false, false);
        }

        // lane-local row max over this wave's 256 columns
        float lm[8];
#pragma unroll
        for (int r = 0; r < 8; ++r) lm[r] = -3.4e38f;
#pragma unroll
        for (int t = 0; t < 16; ++t)
#pragma unroll
            for (int r = 0; r < 8; ++r) {
                const float s = c[t][r] * scale;
                c[t][r] = s;
                lm[r]   = fmaxf(lm[r], s);
            }
        // reduce across the 16 lanes of each half (xor<16 stays within halves)
#pragma unroll
        for (int off = 1; off < 16; off <<= 1)
#pragma unroll
            for (int r = 0; r < 8; ++r) lm[r] = fmaxf(lm[r], __shfl_xor(lm[r], off, 32));
        if (col == 0)
#pragma unroll
            for (int r = 0; r < 8; ++r) redmax[wave * 16 + half + r] = lm[r];
        __syncthreads();
        if (tid < 16) {
            float m = redmax[tid];
            for (int w = 1; w < 8; ++w) m = fmaxf(m, redmax[w * 16 + tid]);
            rowmax[tid] = m;
        }
        __syncthreads();

        float gm[8], ls[8];
#pragma unroll
        for (int r = 0; r < 8; ++r) { gm[r] = rowmax[half + r]; ls[r] = 0.f; }
#pragma unroll
        for (int t = 0; t < 16; ++t)
#pragma unroll
            for (int r = 0; r < 8; ++r) {
                const float e = __expf(c[t][r] - gm[r]);
                c[t][r] = e;
                ls[r]  += e;
            }
#pragma unroll
        for (int off = 1; off < 16; off <<= 1)
#pragma unroll
            for (int r = 0; r < 8; ++r) ls[r] += __shfl_xor(ls[r], off, 32);
        if (col == 0)
#pragma unroll
            for (int r = 0; r < 8; ++r) redsum[wave * 16 + half + r] = ls[r];
        __syncthreads();
        if (tid < 16) {
            float s = 0.f;
            for (int w = 0; w < 8; ++w) s += redsum[w * 16 + tid];
            rowsum[tid] = s;
        }
        __syncthreads();

        float gs[8];
#pragma unroll
        for (int r = 0; r < 8; ++r) gs[r] = 1.0f / rowsum[half + r];
#pragma unroll
        for (int t = 0; t < 16; ++t)
#pragma unroll
            for (int r = 0; r < 8; ++r) {
                const int row = half + r;
                const int n   = wave * 256 + t * 16 + col;
                acc[row * ACC_STRIDE + n] += c[t][r] * gs[r];   // unique owner per (row,n)
            }
        __syncthreads();
    }

    // ---- Gaussian temporal-distance reweight + renormalize ----
    {
        const int m  = tid >> 4;   // row 0..15
        const int c0 = tid & 15;   // 16 threads per row
        const int qi = q0 + m;
        float psum = 0.f;
        for (int n = c0; n < L_; n += 16) {
            const float dd = (float)(qi - n);
            const float w  = acc[m * ACC_STRIDE + n] * __expf(-0.5f * dd * dd);
            acc[m * ACC_STRIDE + n] = w;
            psum += w;
        }
        redmax[m * 16 + c0] = psum;   // reuse scratch as [16][16]
        __syncthreads();
        float rs = 0.f;
#pragma unroll
        for (int j = 0; j < 16; ++j) rs += redmax[m * 16 + j];
        const float inv  = 1.0f / rs;
        float*      wrow = weightedp + ((size_t)b * L_ + qi) * L_;
        for (int n = c0; n < L_; n += 16) {
            const float w = acc[m * ACC_STRIDE + n] * inv;
            __builtin_nontemporal_store(w, wrow + n);        // output #2 (streaming)
            awb[m * AWB_STRIDE + n] = (bf16_t)w;             // bf16 A for final GEMM
        }
    }
    __syncthreads();

    // ---- out[16][512] = weighted[16][2048] @ v_b[2048][512] (via vt) ----
    {
        v8f o[4];
#pragma unroll
        for (int t = 0; t < 4; ++t) o[t] = vzero8();
        const bf16_t* vtb  = vt + (size_t)b * D_ * L_;
        const bf16_t* arow = awb + (size_t)col * AWB_STRIDE;
        for (int kc = 0; kc < L_ / 32; ++kc) {
            v16bf a = load_a_frag(arow, kc, lane);          // ds_load_b128 x2, conflict-free
#pragma unroll
            for (int t = 0; t < 4; ++t) {
                const int n = wave * 64 + t * 16 + col;
                v16bf bb = load_b_frag(vtb + (size_t)n * L_, kc, lane);
                o[t] = __builtin_amdgcn_wmma_f32_16x16x32_bf16(false, a, false, bb,
                                                               (short)0, o[t], false, false);
            }
        }
#pragma unroll
        for (int t = 0; t < 4; ++t) {
            const int n = wave * 64 + t * 16 + col;
#pragma unroll
            for (int r = 0; r < 8; ++r) {
                const int M = r + half;
                __builtin_nontemporal_store(o[t][r],
                    outp + ((size_t)(q0 + M) * B_ + b) * D_ + n);   // output #1 [L,B,D]
            }
        }
    }
}

// ---------------------------------------------------------------------------
extern "C" void kernel_launch(void* const* d_in, const int* in_sizes, int n_in,
                              void* d_out, int out_size, void* d_ws, size_t ws_size,
                              hipStream_t stream) {
    (void)in_sizes; (void)n_in; (void)out_size; (void)ws_size;
    const float* q  = (const float*)d_in[0];
    const float* k  = (const float*)d_in[1];
    const float* v  = (const float*)d_in[2];
    const float* Wq = (const float*)d_in[3];
    const float* bq = (const float*)d_in[4];
    const float* Wk = (const float*)d_in[5];
    const float* bk = (const float*)d_in[6];

    float* outp      = (float*)d_out;
    float* weightedp = outp + (size_t)L_ * B_ * D_;

    // workspace: Qh | Kh | vt | qb | kb | Wqb | Wkb  (all bf16, ~41 MB)
    bf16_t* Qh  = (bf16_t*)d_ws;
    bf16_t* Kh  = Qh  + (size_t)B_ * H_ * L_ * DK_;
    bf16_t* vt  = Kh  + (size_t)B_ * H_ * L_ * DK_;
    bf16_t* qb  = vt  + (size_t)B_ * D_ * L_;
    bf16_t* kb  = qb  + (size_t)LB_ * D_;
    bf16_t* Wqb = kb  + (size_t)LB_ * D_;
    bf16_t* Wkb = Wqb + (size_t)D_ * D_;

    // Stage 0: downcast operands once (removes all cvt work from GEMM loops)
    cvt_kernel<<<dim3((LB_ * D_) / (256 * 8)), 256, 0, stream>>>(q, qb);
    cvt_kernel<<<dim3((LB_ * D_) / (256 * 8)), 256, 0, stream>>>(k, kb);
    cvt_kernel<<<dim3((D_ * D_) / (256 * 8)), 256, 0, stream>>>(Wq, Wqb);
    cvt_kernel<<<dim3((D_ * D_) / (256 * 8)), 256, 0, stream>>>(Wk, Wkb);
    vt_kernel<<<dim3(L_ / 8, B_), D_, 0, stream>>>(v, vt);

    // Stage 1: projections (bf16 WMMA GEMMs)
    proj_kernel<<<dim3(LB_ / 32), 256, 0, stream>>>(qb, Wqb, bq, Qh);
    proj_kernel<<<dim3(LB_ / 32), 256, 0, stream>>>(kb, Wkb, bk, Kh);

    // Stage 2: fused scores -> softmax -> head-avg -> dist reweight -> out GEMM
    attn_kernel<<<dim3(L_ / 16, B_), 256, SMEM_BYTES, stream>>>(Qh, Kh, vt, outp, weightedp);
}
